// SelfAttention_26645977104514
// MI455X (gfx1250) — matrix operand order repaired and checked by reading, not compiled, not run on hardware
//
#include <hip/hip_runtime.h>
#include <hip/hip_bf16.h>

// ---------------------------------------------------------------------------
// CDNA5 WMMA helpers (wave32, v_wmma_f32_16x16x32_f16)
// ---------------------------------------------------------------------------
typedef __attribute__((ext_vector_type(16))) _Float16 v16h;
typedef __attribute__((ext_vector_type(8)))  float    v8f;

struct alignas(16) H8 { _Float16 h[8]; };

static __device__ __forceinline__ v16h load_frag(const _Float16* p0, const _Float16* p1) {
  v16h r;
  H8* d = reinterpret_cast<H8*>(&r);
  d[0] = *reinterpret_cast<const H8*>(p0);
  d[1] = *reinterpret_cast<const H8*>(p1);
  return r;
}

static __device__ __forceinline__ v8f wmma_f16(v16h a, v16h b, v8f c) {
  return __builtin_amdgcn_wmma_f32_16x16x32_f16(false, a, false, b, (short)0, c, false, false);
}

// LDS byte offset of a __shared__ object: generic LDS pointers carry the
// wave-relative LDS address in addr[31:0] (ISA aperture rules).
static __device__ __forceinline__ unsigned lds_off(const void* p) {
  return (unsigned)(uintptr_t)p;
}

// Async 32-byte global->LDS copy (CDNA5 ASYNCcnt path). INST_OFFSET applies to
// both LDS and memory addresses, so the second half is offset:16 on same regs.
static __device__ __forceinline__ void async_copy32(unsigned lds_byte, const void* gptr) {
  asm volatile("global_load_async_to_lds_b128 %0, %1, off"
               :: "v"(lds_byte), "v"(gptr) : "memory");
  asm volatile("global_load_async_to_lds_b128 %0, %1, off offset:16"
               :: "v"(lds_byte), "v"(gptr) : "memory");
}
static __device__ __forceinline__ void wait_async0() {
  asm volatile("s_wait_asynccnt 0x0" ::: "memory");
}

// ---------------------------------------------------------------------------
// 16-element register staging (load global -> regs, convert, packed LDS store)
// ---------------------------------------------------------------------------
template <typename T> struct Stage16;

template <> struct Stage16<float> {
  float4 f[4];
  __device__ __forceinline__ void load(const float* g) {
    const float4* p = reinterpret_cast<const float4*>(g);
    f[0] = p[0]; f[1] = p[1]; f[2] = p[2]; f[3] = p[3];
  }
  __device__ __forceinline__ void store(_Float16* s) const {
    H8 a, b;
    a.h[0] = (_Float16)f[0].x; a.h[1] = (_Float16)f[0].y;
    a.h[2] = (_Float16)f[0].z; a.h[3] = (_Float16)f[0].w;
    a.h[4] = (_Float16)f[1].x; a.h[5] = (_Float16)f[1].y;
    a.h[6] = (_Float16)f[1].z; a.h[7] = (_Float16)f[1].w;
    b.h[0] = (_Float16)f[2].x; b.h[1] = (_Float16)f[2].y;
    b.h[2] = (_Float16)f[2].z; b.h[3] = (_Float16)f[2].w;
    b.h[4] = (_Float16)f[3].x; b.h[5] = (_Float16)f[3].y;
    b.h[6] = (_Float16)f[3].z; b.h[7] = (_Float16)f[3].w;
    *reinterpret_cast<H8*>(s)     = a;   // ds_store_b128
    *reinterpret_cast<H8*>(s + 8) = b;   // ds_store_b128
  }
};

template <> struct Stage16<_Float16> {
  H8 a, b;
  __device__ __forceinline__ void load(const _Float16* g) {
    const H8* p = reinterpret_cast<const H8*>(g);
    a = p[0]; b = p[1];
  }
  __device__ __forceinline__ void store(_Float16* s) const {
    *reinterpret_cast<H8*>(s)     = a;
    *reinterpret_cast<H8*>(s + 8) = b;
  }
};

// ---------------------------------------------------------------------------
// GEMM: C[M,N] = A[M,K] @ W[N,K]^T + bias  (torch Linear layout)
// 128x128 tile / 256 threads / 8 waves (wave = 32x64), double-buffered LDS.
// ---------------------------------------------------------------------------
template <typename TA, typename TO, bool SANITIZE>
__global__ __launch_bounds__(256) void gemm_xWT_kernel(
    const TA* __restrict__ A, const float* __restrict__ W,
    const float* __restrict__ bias, TO* __restrict__ C,
    int M, int N, int K) {
  constexpr int BK = 32;
  __shared__ _Float16 sA[2][128 * BK];  // 2 x 8 KB
  __shared__ _Float16 sB[2][128 * BK];  // 2 x 8 KB

  const int tid  = threadIdx.x;
  const int lane = tid & 31;
  const int wave = tid >> 5;
  const int wm   = (wave >> 1) * 32;
  const int wn   = (wave & 1) * 64;
  const int tM   = blockIdx.y * 128;
  const int tN   = blockIdx.x * 128;

  const int srow = tid >> 1;          // staging: 0..127
  const int scol = (tid & 1) * 16;    // 0 or 16

  const TA*    Ag = A + (size_t)(tM + srow) * K + scol;
  const float* Wg = W + (size_t)(tN + srow) * K + scol;
  _Float16* sAs[2] = { &sA[0][srow * BK + scol], &sA[1][srow * BK + scol] };
  _Float16* sBs[2] = { &sB[0][srow * BK + scol], &sB[1][srow * BK + scol] };

  v8f acc[2][4];
#pragma unroll
  for (int i = 0; i < 2; ++i)
#pragma unroll
    for (int j = 0; j < 4; ++j) acc[i][j] = {};

  const int ar = lane & 15;
  const int ak = (lane >> 4) * 8;
  const int bk = (lane >> 4) * 8;

  // prologue: stage tile 0
  {
    Stage16<TA> stA; Stage16<float> stW;
    stA.load(Ag); stW.load(Wg);
    stA.store(sAs[0]); stW.store(sBs[0]);
  }
  __syncthreads();

  for (int k0 = 0; k0 < K; k0 += BK) {
    const int cur  = (k0 / BK) & 1;
    const int nxt  = cur ^ 1;
    const bool more = (k0 + BK) < K;

    // issue next tile's global loads before compute (latency overlap)
    Stage16<TA> stA; Stage16<float> stW;
    if (more) { stA.load(Ag + k0 + BK); stW.load(Wg + k0 + BK); }

    v16h aF[2], bF[4];
#pragma unroll
    for (int mi = 0; mi < 2; ++mi) {
      const _Float16* p = &sA[cur][(wm + mi * 16 + ar) * BK + ak];
      aF[mi] = load_frag(p, p + 16);
    }
#pragma unroll
    for (int ni = 0; ni < 4; ++ni) {
      const _Float16* p = &sB[cur][(wn + ni * 16 + ar) * BK + bk];
      bF[ni] = load_frag(p, p + 16);
    }
#pragma unroll
    for (int mi = 0; mi < 2; ++mi)
#pragma unroll
      for (int ni = 0; ni < 4; ++ni)
        acc[mi][ni] = wmma_f16(aF[mi], bF[ni], acc[mi][ni]);

    if (more) { stA.store(sAs[nxt]); stW.store(sBs[nxt]); }
    __syncthreads();
  }

  // C layout: VGPR r, lane l -> row = r + 8*(l>>4), col = l&15
  const int rhalf = (lane >> 4) * 8;
  const int cl    = lane & 15;
#pragma unroll
  for (int mi = 0; mi < 2; ++mi)
#pragma unroll
    for (int ni = 0; ni < 4; ++ni) {
      const int col = tN + wn + ni * 16 + cl;
      const float bv = bias[col];
#pragma unroll
      for (int r = 0; r < 8; ++r) {
        const int row = tM + wm + mi * 16 + rhalf + r;
        float v = acc[mi][ni][r] + bv;
        if (SANITIZE) v = (v != v) ? 0.0f : fminf(fmaxf(v, -1e9f), 1e9f);
        C[(size_t)row * N + col] = (TO)v;
      }
    }
}

// ---------------------------------------------------------------------------
// Flash attention: one block per (b*h, 128 query rows); 8 waves x 16 rows.
// K tile staged via async global->LDS (ASYNCcnt); V staged transposed with
// packed b32 stores (key pairs contiguous in sVT rows).
// ---------------------------------------------------------------------------
__global__ __launch_bounds__(256) void attention_kernel(
    const _Float16* __restrict__ Qh, const _Float16* __restrict__ Kh,
    const _Float16* __restrict__ Vh, _Float16* __restrict__ Ctx,
    int Bsz, int S, int E, int H, int D) {
  __shared__ _Float16 sK[32 * 128];     // 8 KB: key x d
  __shared__ _Float16 sVT[128 * 32];    // 8 KB: d x key
  __shared__ _Float16 sP[8][16 * 32];   // 8 KB: per-wave P scratch

  const int bh = blockIdx.x;
  const int b = bh / H, h = bh % H;
  const int tid = threadIdx.x, lane = tid & 31, wave = tid >> 5;
  const int q0 = blockIdx.y * 128 + wave * 16;

  const int ar = lane & 15;
  const int ak = (lane >> 4) * 8;
  const int bk = (lane >> 4) * 8;

  // Q A-fragments: 16 rows, D=128 -> 4 K-slices of 32
  v16h qF[4];
#pragma unroll
  for (int s = 0; s < 4; ++s) {
    const _Float16* p = Qh + (size_t)(b * S + q0 + ar) * E + h * D + s * 32 + ak;
    qF[s] = load_frag(p, p + 16);
  }

  v8f o[8];
#pragma unroll
  for (int t = 0; t < 8; ++t) o[t] = {};
  float mrow[8], lsum[8];
#pragma unroll
  for (int r = 0; r < 8; ++r) { mrow[r] = -1e30f; lsum[r] = 0.0f; }

  const float scale = 0.08838834764831845f;  // 1/sqrt(128)

  // K staging coords: 32 keys x 128 d -> 256 threads x 16 halves (async)
  const int krow = tid >> 3;
  const int kcol = (tid & 7) * 16;
  const unsigned sK_dst = lds_off(&sK[krow * 128 + kcol]);
  // V staging coords: key pairs x 8-d chunks -> packed b32 transposed stores
  const int vkey = (tid >> 4) * 2;       // 0,2,..,30
  const int vdch = (tid & 15) * 8;       // 0,8,..,120

  for (int kb = 0; kb < S; kb += 32) {
    // async stage K tile (raw f16 bytes, no VGPR roundtrip)
    async_copy32(sK_dst, Kh + (size_t)(b * S + kb + krow) * E + h * D + kcol);

    // stage V transposed: two adjacent keys packed per b32 store
    {
      const _Float16* v0 = Vh + (size_t)(b * S + kb + vkey) * E + h * D + vdch;
      H8 r0 = *reinterpret_cast<const H8*>(v0);
      H8 r1 = *reinterpret_cast<const H8*>(v0 + E);
#pragma unroll
      for (int j = 0; j < 8; ++j) {
        union { _Float16 h[2]; unsigned u; } pk;
        pk.h[0] = r0.h[j]; pk.h[1] = r1.h[j];
        *reinterpret_cast<unsigned*>(&sVT[(vdch + j) * 32 + vkey]) = pk.u;
      }
    }
    wait_async0();
    __syncthreads();

    // scores: two 16x16 tiles (keys kb..+15, kb+16..+31), 8 WMMAs
    v8f s0 = {}, s1 = {};
#pragma unroll
    for (int s = 0; s < 4; ++s) {
      const _Float16* p0 = &sK[ar * 128 + s * 32 + bk];
      const _Float16* p1 = &sK[(16 + ar) * 128 + s * 32 + bk];
      s0 = wmma_f16(qF[s], load_frag(p0, p0 + 16), s0);
      s1 = wmma_f16(qF[s], load_frag(p1, p1 + 16), s1);
    }

    // online softmax on C-layout frags (16-lane shuffle reductions per half)
#pragma unroll
    for (int r = 0; r < 8; ++r) {
      float v0 = s0[r] * scale;
      float v1 = s1[r] * scale;
      float mx = fmaxf(v0, v1);
#pragma unroll
      for (int off = 1; off < 16; off <<= 1)
        mx = fmaxf(mx, __shfl_xor(mx, off, 32));
      const float mnew = fmaxf(mrow[r], mx);
      const float corr = __expf(mrow[r] - mnew);
      const float p0 = __expf(v0 - mnew);
      const float p1 = __expf(v1 - mnew);
      float rs = p0 + p1;
#pragma unroll
      for (int off = 1; off < 16; off <<= 1)
        rs += __shfl_xor(rs, off, 32);
      lsum[r] = lsum[r] * corr + rs;
      mrow[r] = mnew;
      s0[r] = p0;
      s1[r] = p1;
#pragma unroll
      for (int t = 0; t < 8; ++t) o[t][r] *= corr;
    }

    // spill P (16x32) C-layout -> row-major LDS, reload as A-fragment
    {
      const int prow = (lane >> 4) * 8;
      const int pc = lane & 15;
#pragma unroll
      for (int r = 0; r < 8; ++r) {
        union { _Float16 h[2]; unsigned u; } pk;
        pk.h[0] = (_Float16)s0[r]; pk.h[1] = (_Float16)0.0f;
        sP[wave][(prow + r) * 32 + pc]      = pk.h[0];
        sP[wave][(prow + r) * 32 + pc + 16] = (_Float16)s1[r];
      }
    }
    const _Float16* pp = &sP[wave][ar * 32 + ak];
    v16h pF = load_frag(pp, pp + 16);  // DS in-order within wave

    // P @ V: 8 WMMAs over the 8 d-tiles
#pragma unroll
    for (int t = 0; t < 8; ++t) {
      const _Float16* vp = &sVT[(t * 16 + ar) * 32 + bk];
      o[t] = wmma_f16(pF, load_frag(vp, vp + 16), o[t]);
    }
    __syncthreads();
  }

  const int rhalf = (lane >> 4) * 8;
  const int cl = lane & 15;
#pragma unroll
  for (int t = 0; t < 8; ++t)
#pragma unroll
    for (int r = 0; r < 8; ++r) {
      const int row = q0 + rhalf + r;
      const float val = o[t][r] / lsum[r];
      Ctx[(size_t)(b * S + row) * E + h * D + t * 16 + cl] = (_Float16)val;
    }
}

// ---------------------------------------------------------------------------
// LayerNorm over last dim (in-place safe)
// ---------------------------------------------------------------------------
__global__ __launch_bounds__(256) void layernorm_kernel(
    const float* __restrict__ X, const float* __restrict__ gamma,
    const float* __restrict__ beta, float* __restrict__ Y, int N) {
  __shared__ float sred[2][8];
  const int row = blockIdx.x;
  const int tid = threadIdx.x, lane = tid & 31, wave = tid >> 5;
  const float* x = X + (size_t)row * N;
  float s = 0.f, ss = 0.f;
  for (int i = tid; i < N; i += 256) {
    float v = x[i];
    s += v; ss += v * v;
  }
#pragma unroll
  for (int off = 1; off < 32; off <<= 1) {
    s  += __shfl_xor(s, off, 32);
    ss += __shfl_xor(ss, off, 32);
  }
  if (lane == 0) { sred[0][wave] = s; sred[1][wave] = ss; }
  __syncthreads();
  if (wave == 0) {
    s  = (lane < 8) ? sred[0][lane] : 0.f;
    ss = (lane < 8) ? sred[1][lane] : 0.f;
#pragma unroll
    for (int off = 1; off < 8; off <<= 1) {
      s  += __shfl_xor(s, off, 32);
      ss += __shfl_xor(ss, off, 32);
    }
    if (lane == 0) { sred[0][0] = s; sred[1][0] = ss; }
  }
  __syncthreads();
  const float mu   = sred[0][0] / N;
  const float var  = sred[1][0] / N - mu * mu;
  const float rstd = rsqrtf(var + 1e-5f);
  float* y = Y + (size_t)row * N;
  for (int i = tid; i < N; i += 256)
    y[i] = (x[i] - mu) * rstd * gamma[i] + beta[i];
}

// ---------------------------------------------------------------------------
// Launcher
// ---------------------------------------------------------------------------
extern "C" void kernel_launch(void* const* d_in, const int* in_sizes, int n_in,
                              void* d_out, int out_size, void* d_ws, size_t ws_size,
                              hipStream_t stream) {
  const float* x     = (const float*)d_in[0];
  const float* Wq    = (const float*)d_in[1];
  const float* bq    = (const float*)d_in[2];
  const float* Wk    = (const float*)d_in[3];
  const float* bk    = (const float*)d_in[4];
  const float* Wv    = (const float*)d_in[5];
  const float* bv    = (const float*)d_in[6];
  const float* Wo    = (const float*)d_in[7];
  const float* bo    = (const float*)d_in[8];
  const float* gamma = (const float*)d_in[9];
  const float* beta  = (const float*)d_in[10];

  const int B = 4, S = 1024, E = 2048, H = 16, D = 128;
  const int M = B * S;

  _Float16* Qh   = (_Float16*)d_ws;
  _Float16* Kh   = Qh + (size_t)M * E;
  _Float16* Vh   = Kh + (size_t)M * E;
  _Float16* Ctxh = Vh + (size_t)M * E;
  float*    Of   = (float*)d_out;  // o-proj writes fp32 output; LN runs in place

  dim3 blk(256);
  dim3 ggemm(E / 128, M / 128);

  gemm_xWT_kernel<float, _Float16, true ><<<ggemm, blk, 0, stream>>>(x, Wq, bq, Qh, M, E, E);
  gemm_xWT_kernel<float, _Float16, true ><<<ggemm, blk, 0, stream>>>(x, Wk, bk, Kh, M, E, E);
  gemm_xWT_kernel<float, _Float16, false><<<ggemm, blk, 0, stream>>>(x, Wv, bv, Vh, M, E, E);

  attention_kernel<<<dim3(B * H, S / 128), blk, 0, stream>>>(Qh, Kh, Vh, Ctxh, B, S, E, H, D);

  gemm_xWT_kernel<_Float16, float, false><<<ggemm, blk, 0, stream>>>(Ctxh, Wo, bo, Of, M, E, E);

  layernorm_kernel<<<dim3(M), blk, 0, stream>>>(Of, gamma, beta, (float*)d_out, E);
}
